// DevignGGNN_51677046505790
// MI455X (gfx1250) — matrix operand order repaired and checked by reading, not compiled, not run on hardware
//
#include <hip/hip_runtime.h>
#include <hip/hip_bf16.h>

// ---------------------------------------------------------------------------
// Problem constants (match the JAX reference)
// ---------------------------------------------------------------------------
#define NB       32
#define NPG_     512
#define HIDC     256
#define NNODES   16384            // NB * NPG_
#define NETYPES  13
#define NEDGES   262144
#define NSTEPS   6
#define CATC     512              // HID + IN_DIM

typedef __attribute__((ext_vector_type(16))) __bf16 v16bf;
typedef __attribute__((ext_vector_type(8)))  __bf16 v8bf;
typedef __attribute__((ext_vector_type(8)))  float  v8f;

__device__ __forceinline__ __bf16 f2bf(float f) {
  unsigned int u = __builtin_bit_cast(unsigned int, f);
  u += 0x7FFFu + ((u >> 16) & 1u);               // round-to-nearest-even
  unsigned short s = (unsigned short)(u >> 16);
  return __builtin_bit_cast(__bf16, s);
}

// ---------------------------------------------------------------------------
// Pack a (K x Nc) fp32 matrix (arbitrary strides, so transposes/conv taps are
// free) into WMMA-B fragment-major bf16: per 32x16 K-tile, lane L holds 16
// contiguous bf16 (column n = L&15, K-half selected by L>>4).
// grid.x = (K/32)*(Nc/16) tiles, grid.z = taps/etypes, block = 32 lanes.
// ---------------------------------------------------------------------------
__global__ void pack_b_bf16(const float* __restrict__ src, long sk, long sn,
                            long srcZ, __bf16* __restrict__ dst, long dstZ,
                            int ntilesN) {
  const int tile = blockIdx.x;
  const int lane = threadIdx.x;
  const int hf = lane >> 4, r = lane & 15;
  const int kt = tile / ntilesN, nt = tile % ntilesN;
  const float* s = src + (long)blockIdx.z * srcZ;
  __bf16* d = dst + (long)blockIdx.z * dstZ + ((long)tile * 32 + lane) * 16;
  const int n = nt * 16 + r;
  const int kbase = kt * 32 + (hf ? 16 : 0);
#pragma unroll
  for (int e = 0; e < 16; ++e)
    d[e] = f2bf(s[(long)(kbase + e) * sk + (long)n * sn]);
}

// ---------------------------------------------------------------------------
// WMMA bf16 GEMM:  C[M x Nc] (+bias, opt relu) = sum_t A[rows+t] * Bpacked[t]
// A: row-major bf16 (lda), rows clamped to Arows (conv tap overrun guard).
//   The 16xK A tile is staged once per block into LDS via CDNA5 async
//   global->LDS DMA loads (ASYNCcnt path), then shared by all 4 waves.
// B: fragment-packed (pack_b_bf16), tap stride = (K/32)*(Nc/16)*512 elems.
// One wave per 16x16 C tile; 4 waves per block; grid = (M/16, Nc/64).
// All grids tile M and Nc exactly -> no divergence guards needed anywhere.
// ---------------------------------------------------------------------------
__global__ void __launch_bounds__(128)
wmma_gemm_bf16(const __bf16* __restrict__ A, int lda, int Arows,
               const __bf16* __restrict__ Bp,
               const float* __restrict__ bias,
               float* __restrict__ C, int ldc,
               int Nc, int K, int ntaps, int relu) {
  __shared__ __bf16 shA[16 * 512];            // 16 KB: 16 rows x K<=512 bf16

  const int tid  = threadIdx.x;
  const int lane = tid & 31;
  const int wave = tid >> 5;
  const int hf = lane >> 4;
  const int r  = lane & 15;
  const int m0 = blockIdx.x * 16;
  const int n0 = (blockIdx.y * 4 + wave) * 16;
  const int ktiles = K >> 5;
  const int ntilesN = Nc >> 4;
  const long tapB  = (long)ktiles * ntilesN * 512;
  const long bstep = (long)ntilesN * 512;
  // 16B chunks per A row: always a power of two (32 or 64) -> shift/mask
  const int chunksPerRow = K >> 3;
  const int cshift = __builtin_ctz(chunksPerRow);
  const int cmask  = chunksPerRow - 1;
  const int reps   = (chunksPerRow << 4) >> 7; // totalChunks/128 (uniform)

  v8f acc = {0.f, 0.f, 0.f, 0.f, 0.f, 0.f, 0.f, 0.f};

  for (int t = 0; t < ntaps; ++t) {
    // ---- stage A tile into LDS with async global->LDS loads (ASYNCcnt) ----
    for (int rep = 0; rep < reps; ++rep) {
      const int q   = rep * 128 + tid;
      const int row = q >> cshift;
      const int cc  = q & cmask;
      long srow = (long)m0 + row + t;
      if (srow >= Arows) srow = Arows - 1;    // padded tap rows: unused result
      const __bf16* gp = A + srow * (long)lda + cc * 8;
      const unsigned long long ga = (unsigned long long)(size_t)gp;
      // LDS flat addresses carry the LDS byte offset in addr[31:0]
      const unsigned ldsb = (unsigned)(size_t)(&shA[(row << cshift << 3) + cc * 8]);
      asm volatile("global_load_async_to_lds_b128 %0, %1, off"
                   :: "v"(ldsb), "v"(ga) : "memory");
    }
    asm volatile("s_wait_asynccnt 0x0" ::: "memory");
    __syncthreads();                          // publish LDS tile to all waves

    // ---- K loop: A fragments from LDS, packed B from global, WMMA ----
    const __bf16* ap = &shA[r * K + (hf ? 16 : 0)];
    const __bf16* bp = Bp + (long)t * tapB + ((long)(n0 >> 4) * 32 + lane) * 16;
    for (int kk = 0; kk < ktiles; ++kk, ap += 32, bp += bstep) {
      v8bf alo = *(const v8bf*)ap;            // ds_load_b128
      v8bf ahi = *(const v8bf*)(ap + 8);
      v16bf afrag;
#pragma unroll
      for (int i = 0; i < 8; ++i) { afrag[i] = alo[i]; afrag[8 + i] = ahi[i]; }
      v16bf bfrag = *(const v16bf*)bp;
      if (kk + 1 < ktiles)                    // keep packed-B stream warm
        __builtin_prefetch(bp + bstep, 0, 3);
      acc = __builtin_amdgcn_wmma_f32_16x16x32_bf16(
                false, afrag, false, bfrag, (short)0, acc, false, false);
    }
    __syncthreads();                          // tile consumed; safe to restage
  }

  const int col = n0 + r;
  const float bv = bias ? bias[col] : 0.f;
#pragma unroll
  for (int j = 0; j < 8; ++j) {               // C layout: VGPR j -> row m0+j(+8)
    const int rowc = m0 + j + (hf ? 8 : 0);   // always < M (exact tiling)
    float v = acc[j] + bv;
    if (relu) v = fmaxf(v, 0.f);
    C[(long)rowc * ldc + col] = v;
  }
}

// ---------------------------------------------------------------------------
// Elementwise helpers
// ---------------------------------------------------------------------------
__global__ void cast_f32_bf16(const float* __restrict__ s, __bf16* __restrict__ d,
                              long n) {
  long i = (long)blockIdx.x * blockDim.x + threadIdx.x;
  if (i < n) d[i] = f2bf(s[i]);
}

__global__ void zero_f32(float* __restrict__ p, long n) {
  long i = (long)blockIdx.x * blockDim.x + threadIdx.x;
  if (i < n) p[i] = 0.f;
}

__global__ void concat_bf16(const float* __restrict__ h, const float* __restrict__ x,
                            __bf16* __restrict__ d) {
  long i = (long)blockIdx.x * blockDim.x + threadIdx.x;
  if (i >= (long)NNODES * CATC) return;
  const int row = (int)(i >> 9);
  const int j = (int)(i & (CATC - 1));
  const float v = (j < HIDC) ? h[(long)row * HIDC + j]
                             : x[(long)row * HIDC + (j - HIDC)];
  d[i] = f2bf(v);
}

// gather msg = tn_k[src], scatter-add into a[dst] for edges of etype k.
// 64 threads / edge, 4 floats each (tn_k + a are L2-resident: 16MB each).
__global__ void scatter_etype(const float* __restrict__ tnk,
                              const int* __restrict__ src,
                              const int* __restrict__ dst,
                              const int* __restrict__ et, int k,
                              float* __restrict__ a) {
  const long idx = (long)blockIdx.x * blockDim.x + threadIdx.x;
  const int e = (int)(idx >> 6);
  if (e >= NEDGES) return;
  if (et[e] != k) return;
  const int c = ((int)idx & 63) * 4;
  const float4 m = *(const float4*)(tnk + (long)src[e] * HIDC + c);
  float* o = a + (long)dst[e] * HIDC + c;
  unsafeAtomicAdd(o + 0, m.x);
  unsafeAtomicAdd(o + 1, m.y);
  unsafeAtomicAdd(o + 2, m.z);
  unsafeAtomicAdd(o + 3, m.w);
}

__global__ void gru_update(const float* __restrict__ gi, const float* __restrict__ gh,
                           float* __restrict__ h) {
  const long i = (long)blockIdx.x * blockDim.x + threadIdx.x;
  if (i >= (long)NNODES * HIDC) return;
  const int n = (int)(i >> 8);
  const int c = (int)(i & 255);
  const float* gin = gi + (long)n * 768;
  const float* ghn = gh + (long)n * 768;
  const float ir = gin[c], iz = gin[256 + c], in_ = gin[512 + c];
  const float hr = ghn[c], hz = ghn[256 + c], hn = ghn[512 + c];
  const float rr = 1.f / (1.f + __expf(-(ir + hr)));
  const float zz = 1.f / (1.f + __expf(-(iz + hz)));
  const float nn = tanhf(in_ + rr * hn);
  h[i] = (1.f - zz) * nn + zz * h[i];
}

// maxpool k=3 s=2 over padded L=512 rows (valid l<510) -> 254 per graph
__global__ void pool3(const float* __restrict__ in, float* __restrict__ out, int C) {
  const long i = (long)blockIdx.x * blockDim.x + threadIdx.x;
  const long total = (long)NB * 254 * C;
  if (i >= total) return;
  const int o = (int)(i % C);
  const long t = i / C;
  const int p = (int)(t % 254);
  const int b = (int)(t / 254);
  const float* base = in + ((long)b * NPG_ + 2 * p) * C + o;
  out[((long)b * 254 + p) * C + o] =
      fmaxf(base[0], fmaxf(base[C], base[2 * (long)C]));
}

// maxpool k=2 s=2 over L=254 -> 127 per graph
__global__ void pool2(const float* __restrict__ in, float* __restrict__ out, int C) {
  const long i = (long)blockIdx.x * blockDim.x + threadIdx.x;
  const long total = (long)NB * 127 * C;
  if (i >= total) return;
  const int o = (int)(i % C);
  const long t = i / C;
  const int p = (int)(t % 127);
  const int b = (int)(t / 127);
  const float* base = in + ((long)b * 254 + 2 * p) * C + o;
  out[((long)b * 127 + p) * C + o] = fmaxf(base[0], base[C]);
}

__global__ void final_head(const float* __restrict__ Y2, const float* __restrict__ Z2,
                           const float* __restrict__ wy, const float* __restrict__ by,
                           const float* __restrict__ wz, const float* __restrict__ bz,
                           float* __restrict__ out) {
  __shared__ float red[128];
  const int b = blockIdx.x;
  const int t = threadIdx.x;
  float acc = 0.f;
  if (t < 127) {
    const float* y = Y2 + ((long)b * 127 + t) * HIDC;
    const float* z = Z2 + ((long)b * 127 + t) * CATC;
    float yv = by[0];
    for (int c = 0; c < HIDC; ++c) yv += y[c] * wy[c];
    float zv = bz[0];
    for (int c = 0; c < CATC; ++c) zv += z[c] * wz[c];
    acc = yv * zv;
  }
  red[t] = acc;
  __syncthreads();
  for (int s = 64; s > 0; s >>= 1) {
    if (t < s) red[t] += red[t + s];
    __syncthreads();
  }
  if (t == 0) out[b] = 1.f / (1.f + __expf(-red[0] / 127.f));
}

// ---------------------------------------------------------------------------
// Host orchestration
// ---------------------------------------------------------------------------
extern "C" void kernel_launch(void* const* d_in, const int* in_sizes, int n_in,
                              void* d_out, int out_size, void* d_ws, size_t ws_size,
                              hipStream_t stream) {
  (void)in_sizes; (void)n_in; (void)out_size; (void)ws_size;
  const float* x       = (const float*)d_in[0];
  const int*   src     = (const int*)  d_in[1];
  const int*   dst     = (const int*)  d_in[2];
  const int*   etype   = (const int*)  d_in[3];
  const float* W_e     = (const float*)d_in[4];
  const float* b_e     = (const float*)d_in[5];
  const float* wih     = (const float*)d_in[6];
  const float* whh     = (const float*)d_in[7];
  const float* bih     = (const float*)d_in[8];
  const float* bhh     = (const float*)d_in[9];
  const float* conv1_w = (const float*)d_in[10];
  const float* conv1_b = (const float*)d_in[11];
  const float* conv2_w = (const float*)d_in[12];
  const float* conv2_b = (const float*)d_in[13];
  const float* cc1_w   = (const float*)d_in[14];
  const float* cc1_b   = (const float*)d_in[15];
  const float* cc2_w   = (const float*)d_in[16];
  const float* cc2_b   = (const float*)d_in[17];
  const float* wy      = (const float*)d_in[18];
  const float* by      = (const float*)d_in[19];
  const float* wz      = (const float*)d_in[20];
  const float* bz      = (const float*)d_in[21];
  float* out = (float*)d_out;

  // ---- bump allocator over d_ws ----
  char* wsp = (char*)d_ws;
  auto alloc = [&](size_t bytes) -> void* {
    void* p = (void*)wsp;
    wsp += (bytes + 255) & ~(size_t)255;
    return p;
  };
  float*  h      = (float*) alloc((size_t)NNODES * HIDC * 4);
  __bf16* h_bf   = (__bf16*)alloc((size_t)NNODES * HIDC * 2);
  float*  agg    = (float*) alloc((size_t)NNODES * HIDC * 4);
  __bf16* a_bf   = (__bf16*)alloc((size_t)NNODES * HIDC * 2);
  float*  tnk    = (float*) alloc((size_t)NNODES * HIDC * 4);
  float*  gi     = (float*) alloc((size_t)NNODES * 768 * 4);
  float*  gh     = (float*) alloc((size_t)NNODES * 768 * 4);
  __bf16* We_p   = (__bf16*)alloc((size_t)NETYPES * 65536 * 2);
  __bf16* wih_p  = (__bf16*)alloc((size_t)196608 * 2);
  __bf16* whh_p  = (__bf16*)alloc((size_t)196608 * 2);
  __bf16* c1_p   = (__bf16*)alloc((size_t)3 * 65536 * 2);
  __bf16* c2_p   = (__bf16*)alloc((size_t)65536 * 2);
  __bf16* cc1_p  = (__bf16*)alloc((size_t)3 * 262144 * 2);
  __bf16* cc2_p  = (__bf16*)alloc((size_t)262144 * 2);
  __bf16* cat_bf = (__bf16*)alloc((size_t)NNODES * CATC * 2);
  float*  Y1pre  = (float*) alloc((size_t)NNODES * HIDC * 4);
  float*  P1     = (float*) alloc((size_t)NB * 254 * HIDC * 4);
  __bf16* P1_bf  = (__bf16*)alloc((size_t)NB * 254 * HIDC * 2);
  float*  Y2pre  = (float*) alloc((size_t)NB * 254 * HIDC * 4);
  float*  Y2     = (float*) alloc((size_t)NB * 127 * HIDC * 4);
  float*  Z1pre  = (float*) alloc((size_t)NNODES * CATC * 4);
  float*  PZ1    = (float*) alloc((size_t)NB * 254 * CATC * 4);
  __bf16* PZ1_bf = (__bf16*)alloc((size_t)NB * 254 * CATC * 2);
  float*  Z2pre  = (float*) alloc((size_t)NB * 254 * CATC * 4);
  float*  Z2     = (float*) alloc((size_t)NB * 127 * CATC * 4);

  const long NH = (long)NNODES * HIDC;

  // ---- pack all weight matrices into WMMA-B fragment layout (bf16) ----
  pack_b_bf16<<<dim3(128, 1, NETYPES), 32, 0, stream>>>(W_e, 256, 1, 65536, We_p, 65536, 16);
  pack_b_bf16<<<dim3(384, 1, 1), 32, 0, stream>>>(wih, 1, 256, 0, wih_p, 0, 48);   // B = wih^T
  pack_b_bf16<<<dim3(384, 1, 1), 32, 0, stream>>>(whh, 1, 256, 0, whh_p, 0, 48);   // B = whh^T
  pack_b_bf16<<<dim3(128, 1, 3), 32, 0, stream>>>(conv1_w, 3, 768, 1, c1_p, 65536, 16);
  pack_b_bf16<<<dim3(128, 1, 1), 32, 0, stream>>>(conv2_w, 1, 256, 0, c2_p, 0, 16);
  pack_b_bf16<<<dim3(512, 1, 3), 32, 0, stream>>>(cc1_w, 3, 1536, 1, cc1_p, 262144, 32);
  pack_b_bf16<<<dim3(512, 1, 1), 32, 0, stream>>>(cc2_w, 1, 512, 0, cc2_p, 0, 32);

  // ---- h = x ----
  hipMemcpyAsync(h, x, (size_t)NH * 4, hipMemcpyDeviceToDevice, stream);

  // ---- GGNN: 6 steps ----
  for (int step = 0; step < NSTEPS; ++step) {
    cast_f32_bf16<<<(NH + 255) / 256, 256, 0, stream>>>(h, h_bf, NH);
    zero_f32<<<(NH + 255) / 256, 256, 0, stream>>>(agg, NH);
    for (int k = 0; k < NETYPES; ++k) {
      // tn_k = h @ W_e[k] + b_e[k]   (16 MB result, stays in L2)
      wmma_gemm_bf16<<<dim3(NNODES / 16, 4), 128, 0, stream>>>(
          h_bf, HIDC, NNODES, We_p + (long)k * 65536, b_e + k * HIDC,
          tnk, HIDC, HIDC, HIDC, 1, 0);
      // a[dst] += tn_k[src] for edges of etype k (L2-resident atomics)
      scatter_etype<<<(NEDGES * 64) / 256, 256, 0, stream>>>(tnk, src, dst, etype, k, agg);
    }
    cast_f32_bf16<<<(NH + 255) / 256, 256, 0, stream>>>(agg, a_bf, NH);
    wmma_gemm_bf16<<<dim3(NNODES / 16, 12), 128, 0, stream>>>(
        a_bf, HIDC, NNODES, wih_p, bih, gi, 768, 768, HIDC, 1, 0);
    wmma_gemm_bf16<<<dim3(NNODES / 16, 12), 128, 0, stream>>>(
        h_bf, HIDC, NNODES, whh_p, bhh, gh, 768, 768, HIDC, 1, 0);
    gru_update<<<(NH + 255) / 256, 256, 0, stream>>>(gi, gh, h);
  }

  // ---- readout convs (implicit GEMM: k=3 conv = 3 row-shifted GEMM taps) ----
  cast_f32_bf16<<<(NH + 255) / 256, 256, 0, stream>>>(h, h_bf, NH);
  concat_bf16<<<((long)NNODES * CATC + 255) / 256, 256, 0, stream>>>(h, x, cat_bf);

  wmma_gemm_bf16<<<dim3(NNODES / 16, 4), 128, 0, stream>>>(      // conv1 + relu
      h_bf, HIDC, NNODES, c1_p, conv1_b, Y1pre, HIDC, HIDC, HIDC, 3, 1);
  pool3<<<((long)NB * 254 * HIDC + 255) / 256, 256, 0, stream>>>(Y1pre, P1, HIDC);
  cast_f32_bf16<<<((long)NB * 254 * HIDC + 255) / 256, 256, 0, stream>>>(
      P1, P1_bf, (long)NB * 254 * HIDC);
  wmma_gemm_bf16<<<dim3(NB * 254 / 16, 4), 128, 0, stream>>>(    // conv2 + relu
      P1_bf, HIDC, NB * 254, c2_p, conv2_b, Y2pre, HIDC, HIDC, HIDC, 1, 1);
  pool2<<<((long)NB * 127 * HIDC + 255) / 256, 256, 0, stream>>>(Y2pre, Y2, HIDC);

  wmma_gemm_bf16<<<dim3(NNODES / 16, 8), 128, 0, stream>>>(      // cconv1 + relu
      cat_bf, CATC, NNODES, cc1_p, cc1_b, Z1pre, CATC, CATC, CATC, 3, 1);
  pool3<<<((long)NB * 254 * CATC + 255) / 256, 256, 0, stream>>>(Z1pre, PZ1, CATC);
  cast_f32_bf16<<<((long)NB * 254 * CATC + 255) / 256, 256, 0, stream>>>(
      PZ1, PZ1_bf, (long)NB * 254 * CATC);
  wmma_gemm_bf16<<<dim3(NB * 254 / 16, 8), 128, 0, stream>>>(    // cconv2 + relu
      PZ1_bf, CATC, NB * 254, cc2_p, cc2_b, Z2pre, CATC, CATC, CATC, 1, 1);
  pool2<<<((long)NB * 127 * CATC + 255) / 256, 256, 0, stream>>>(Z2pre, Z2, CATC);

  final_head<<<NB, 128, 0, stream>>>(Y2, Z2, wy, by, wz, bz, out);
}
